// CRNN_61392262529555
// MI455X (gfx1250) — matrix operand order repaired and checked by reading, not compiled, not run on hardware
//
#include <hip/hip_runtime.h>
#include <hip/hip_bf16.h>
#include <math.h>

typedef _Float16 half_t;
typedef __attribute__((ext_vector_type(16))) _Float16 v16h;
typedef __attribute__((ext_vector_type(8)))  _Float16 v8h;
typedef __attribute__((ext_vector_type(8)))  float    v8f;

union V16 { v16h v; v8h h[2]; };

#define BB 32
#define TT 1024
#define MM 40
#define CONV 128
#define HH 256
#define CC 10
#define G3H 768
#define EPSB 1e-5f

__device__ __forceinline__ v8f wmma16(v16h a, v16h b, v8f c) {
  return __builtin_amdgcn_wmma_f32_16x16x32_f16(false, a, false, b, (short)0, c, false, false);
}
__device__ __forceinline__ v8f zero8f() { v8f r = {}; return r; }
__device__ __forceinline__ v16h zero16h() { v16h r = {}; return r; }
__device__ __forceinline__ float sigmoidf_(float x) { return 1.0f / (1.0f + __expf(-x)); }

// ---------------------------------------------------------------------------
// Weight prep: pack conv weights (BN-scale folded) into WMMA B-fragment layout
// dst[tap][kc][nt][lane][j] : n = nt*16 + lane%16 ; k = kc*32 + (lane/16)*8 + (j%8) + (j/8)*16
// ---------------------------------------------------------------------------
__global__ void prep_conv_w(const float* __restrict__ W, const float* __restrict__ b,
                            const float* __restrict__ g, const float* __restrict__ be,
                            const float* __restrict__ m, const float* __restrict__ v,
                            half_t* __restrict__ wB, float* __restrict__ biasOut) {
  int idx = blockIdx.x * 256 + threadIdx.x;
  if (idx >= 25 * 4 * 8 * 32 * 16) return;
  int j    = idx & 15;
  int lane = (idx >> 4) & 31;
  int nt   = (idx >> 9) & 7;
  int kc   = (idx >> 12) & 3;
  int tap  = idx >> 14;
  int n = nt * 16 + (lane & 15);
  int k = kc * 32 + ((lane >> 4) * 8) + (j & 7) + ((j >> 3) * 16);
  int dy = tap / 5, dx = tap % 5;
  float scale = g[n] * rsqrtf(v[n] + EPSB);
  float wv = W[((n * CONV + k) * 5 + dy) * 5 + dx] * scale;
  wB[idx] = (half_t)wv;
  if (idx < CONV) {
    float sc = g[idx] * rsqrtf(v[idx] + EPSB);
    biasOut[idx] = b[idx] * sc + be[idx] - m[idx] * sc;
  }
}

// Generic fp32 [N x srcStride] -> f16 B-fragment pack: dst[kc][nt][lane][j]
__global__ void pack_b_frags(const float* __restrict__ Wsrc, half_t* __restrict__ dst,
                             int nKC, int nNT, int srcStride, int srcColOff, int kLimit,
                             int total) {
  int idx = blockIdx.x * 256 + threadIdx.x;
  if (idx >= total) return;
  int j    = idx & 15;
  int lane = (idx >> 4) & 31;
  int rest = idx >> 9;
  int nt = rest % nNT;
  int kc = rest / nNT;
  int n = nt * 16 + (lane & 15);
  int k = kc * 32 + ((lane >> 4) * 8) + (j & 7) + ((j >> 3) * 16);
  float val = (k < kLimit) ? Wsrc[n * srcStride + srcColOff + k] : 0.0f;
  dst[idx] = (half_t)val;
}

// ---------------------------------------------------------------------------
// Conv1: [B,1,T,40] -> (conv5x5 + BN + ReLU + pool(1,5)) -> f16 [B,T,8,128] CL
// Scalar VALU (K=25, not GEMM-shaped); window staged in LDS.
// ---------------------------------------------------------------------------
__global__ __launch_bounds__(128) void conv1_kernel(
    const float* __restrict__ x, const float* __restrict__ W1, const float* __restrict__ b1,
    const float* __restrict__ g1, const float* __restrict__ be1, const float* __restrict__ m1,
    const float* __restrict__ v1, half_t* __restrict__ out1) {
  int blk = blockIdx.x;              // b*T*8
  int mp = blk & 7;
  int t  = (blk >> 3) & 1023;
  int b  = blk >> 13;
  __shared__ float win[5][9];
  int tid = threadIdx.x;
  if (tid < 45) {
    int r = tid / 9, c = tid % 9;
    int ts = t + r - 2;
    int ms = mp * 5 + c - 2;
    float vv = 0.0f;
    if ((unsigned)ts < (unsigned)TT && (unsigned)ms < (unsigned)MM)
      vv = x[(b * TT + ts) * MM + ms];
    win[r][c] = vv;
  }
  __syncthreads();
  int co = tid;
  float w[25];
#pragma unroll
  for (int i = 0; i < 25; ++i) w[i] = W1[co * 25 + i];
  float scale = g1[co] * rsqrtf(v1[co] + EPSB);
  float bias  = b1[co] * scale + be1[co] - m1[co] * scale;
  float best = 0.0f;  // ReLU floor (max of ReLU == ReLU of max)
#pragma unroll
  for (int pw = 0; pw < 5; ++pw) {
    float acc = 0.0f;
#pragma unroll
    for (int dy = 0; dy < 5; ++dy)
#pragma unroll
      for (int dx = 0; dx < 5; ++dx) acc += w[dy * 5 + dx] * win[dy][pw + dx];
    float y = acc * scale + bias;
    best = fmaxf(best, y);
  }
  out1[((b * TT + t) * 8 + mp) * CONV + co] = (half_t)best;
}

// ---------------------------------------------------------------------------
// Conv2: f16 [B,T,8,128] -> WMMA implicit GEMM (25 taps x K=128) + bias+ReLU
//        + pool(1,4) -> f16 [B,T,2,128]
// Tile: 16 rows = 2 times x 8 mels. Wave = (mtile, 64-wide N half). 8 waves/blk.
// ---------------------------------------------------------------------------
__global__ __launch_bounds__(256) void conv2_kernel(
    const half_t* __restrict__ in, const half_t* __restrict__ wB,
    const float* __restrict__ bias, half_t* __restrict__ out) {
  int wave = threadIdx.x >> 5, lane = threadIdx.x & 31;
  int mt = blockIdx.x * 4 + (wave >> 1);
  int nh = wave & 1;
  int r0 = mt * 16;
  int b  = r0 >> 13;            // 8192 rows per batch
  int t0 = (r0 & 8191) >> 3;    // tile covers t0, t0+1
  int ri = lane & 15;
  int dt = ri >> 3, mel = ri & 7;
  int hf = lane >> 4;
  v8f a0 = zero8f(), a1 = zero8f(), a2 = zero8f(), a3 = zero8f();
  for (int dy = 0; dy < 5; ++dy) {
    int ts = t0 + dt + dy - 2;
    bool tok = (unsigned)ts < (unsigned)TT;
    for (int dx = 0; dx < 5; ++dx) {
      int ms = mel + dx - 2;
      bool ok = tok && ((unsigned)ms < 8u);
      const half_t* arow = in + (size_t)(((b << 10) + ts) * 8 + ms) * CONV;
      int tap = dy * 5 + dx;
      const half_t* bbase = wB + (size_t)tap * 4 * 8 * 512 + (size_t)(nh * 4) * 512 + lane * 16;
#pragma unroll
      for (int kc = 0; kc < 4; ++kc) {
        V16 a;
        if (ok) {
          a.h[0] = *(const v8h*)(arow + kc * 32 + hf * 8);
          a.h[1] = *(const v8h*)(arow + kc * 32 + hf * 8 + 16);
        } else {
          a.v = zero16h();
        }
        const half_t* bb = bbase + (size_t)kc * 8 * 512;
        V16 b0, b1, b2, b3;
        b0.v = *(const v16h*)(bb);
        b1.v = *(const v16h*)(bb + 512);
        b2.v = *(const v16h*)(bb + 1024);
        b3.v = *(const v16h*)(bb + 1536);
        a0 = wmma16(a.v, b0.v, a0);
        a1 = wmma16(a.v, b1.v, a1);
        a2 = wmma16(a.v, b2.v, a2);
        a3 = wmma16(a.v, b3.v, a3);
      }
    }
  }
  // Epilogue: bias + ReLU + maxpool(1,4) over mel. lane half selects time row.
  int tout = t0 + hf;
  v8f accs[4] = {a0, a1, a2, a3};
#pragma unroll
  for (int q = 0; q < 4; ++q) {
    int col = nh * 64 + q * 16 + (lane & 15);
    float bn = bias[col];
    float m0 = -1e30f, m1 = -1e30f;
#pragma unroll
    for (int vv = 0; vv < 4; ++vv) m0 = fmaxf(m0, accs[q][vv] + bn);
#pragma unroll
    for (int vv = 4; vv < 8; ++vv) m1 = fmaxf(m1, accs[q][vv] + bn);
    m0 = fmaxf(m0, 0.0f);
    m1 = fmaxf(m1, 0.0f);
    out[(size_t)(((b << 10) + tout) * 2 + 0) * CONV + col] = (half_t)m0;
    out[(size_t)(((b << 10) + tout) * 2 + 1) * CONV + col] = (half_t)m1;
  }
}

// ---------------------------------------------------------------------------
// Conv3: f16 [B,T,2,128] -> WMMA implicit GEMM + bias+ReLU + pool(1,2)
//        -> feats f16 [(t*32+b), 128]
// Tile: 16 rows = 8 times x 2 mels.
// ---------------------------------------------------------------------------
__global__ __launch_bounds__(256) void conv3_kernel(
    const half_t* __restrict__ in, const half_t* __restrict__ wB,
    const float* __restrict__ bias, half_t* __restrict__ feats) {
  int wave = threadIdx.x >> 5, lane = threadIdx.x & 31;
  int mt = blockIdx.x * 4 + (wave >> 1);
  int nh = wave & 1;
  int r0 = mt * 16;
  int b  = r0 >> 11;            // 2048 rows per batch
  int t0 = (r0 & 2047) >> 1;    // covers t0..t0+7
  int ri = lane & 15;
  int dt = ri >> 1, mel = ri & 1;
  int hf = lane >> 4;
  v8f a0 = zero8f(), a1 = zero8f(), a2 = zero8f(), a3 = zero8f();
  for (int dy = 0; dy < 5; ++dy) {
    int ts = t0 + dt + dy - 2;
    bool tok = (unsigned)ts < (unsigned)TT;
    for (int dx = 0; dx < 5; ++dx) {
      int ms = mel + dx - 2;
      bool ok = tok && ((unsigned)ms < 2u);
      const half_t* arow = in + (size_t)(((b << 10) + ts) * 2 + ms) * CONV;
      int tap = dy * 5 + dx;
      const half_t* bbase = wB + (size_t)tap * 4 * 8 * 512 + (size_t)(nh * 4) * 512 + lane * 16;
#pragma unroll
      for (int kc = 0; kc < 4; ++kc) {
        V16 a;
        if (ok) {
          a.h[0] = *(const v8h*)(arow + kc * 32 + hf * 8);
          a.h[1] = *(const v8h*)(arow + kc * 32 + hf * 8 + 16);
        } else {
          a.v = zero16h();
        }
        const half_t* bb = bbase + (size_t)kc * 8 * 512;
        V16 b0, b1, b2, b3;
        b0.v = *(const v16h*)(bb);
        b1.v = *(const v16h*)(bb + 512);
        b2.v = *(const v16h*)(bb + 1024);
        b3.v = *(const v16h*)(bb + 1536);
        a0 = wmma16(a.v, b0.v, a0);
        a1 = wmma16(a.v, b1.v, a1);
        a2 = wmma16(a.v, b2.v, a2);
        a3 = wmma16(a.v, b3.v, a3);
      }
    }
  }
  v8f accs[4] = {a0, a1, a2, a3};
#pragma unroll
  for (int q = 0; q < 4; ++q) {
    int col = nh * 64 + q * 16 + (lane & 15);
    float bn = bias[col];
#pragma unroll
    for (int p = 0; p < 4; ++p) {
      float mv = fmaxf(accs[q][2 * p] + bn, accs[q][2 * p + 1] + bn);
      mv = fmaxf(mv, 0.0f);
      int t = t0 + hf * 4 + p;
      feats[(size_t)(t * BB + b) * CONV + col] = (half_t)mv;
    }
  }
}

// ---------------------------------------------------------------------------
// giX = feats @ WihX^T + bih : [32768 x 128] x [128 x 768] -> f16 [32768 x 768]
// Wave = (mtile from block, 96-wide N group). 8 waves cover full N.
// ---------------------------------------------------------------------------
__global__ __launch_bounds__(256) void gix_kernel(
    const half_t* __restrict__ feats, const half_t* __restrict__ WihXB,
    const float* __restrict__ bih, half_t* __restrict__ giX) {
  int wave = threadIdx.x >> 5, lane = threadIdx.x & 31;
  int mt = blockIdx.x;
  int hf = lane >> 4;
  int r = mt * 16 + (lane & 15);
  v8f acc[6];
#pragma unroll
  for (int q = 0; q < 6; ++q) acc[q] = zero8f();
  const half_t* arow = feats + (size_t)r * CONV;
#pragma unroll
  for (int kc = 0; kc < 4; ++kc) {
    V16 a;
    a.h[0] = *(const v8h*)(arow + kc * 32 + hf * 8);
    a.h[1] = *(const v8h*)(arow + kc * 32 + hf * 8 + 16);
#pragma unroll
    for (int q = 0; q < 6; ++q) {
      int nt = wave * 6 + q;
      V16 bfr;
      bfr.v = *(const v16h*)(WihXB + (size_t)((kc * 48 + nt) * 32 + lane) * 16);
      acc[q] = wmma16(a.v, bfr.v, acc[q]);
    }
  }
#pragma unroll
  for (int q = 0; q < 6; ++q) {
    int col = (wave * 6 + q) * 16 + (lane & 15);
    float bi = bih[col];
#pragma unroll
    for (int vv = 0; vv < 8; ++vv) {
      int row = mt * 16 + hf * 8 + vv;
      giX[(size_t)row * G3H + col] = (half_t)(acc[q][vv] + bi);
    }
  }
}

// ---------------------------------------------------------------------------
// Persistent GRU scan: one workgroup, 32 waves, 1024 steps.
// Per step: gh = h@Whh^T (8 WMMA k-chunks), gi = giX[t] + tf@WihTf^T (1 chunk).
// r/z tiles in phase1 -> LDS ; n tiles in phase2 combine with r in registers.
// ---------------------------------------------------------------------------
__global__ __launch_bounds__(1024) void gru_kernel(
    const half_t* __restrict__ giX, const half_t* __restrict__ WhhB,
    const half_t* __restrict__ WihTfB, const float* __restrict__ bhh,
    const float* __restrict__ Wf, const float* __restrict__ bfv,
    const float* __restrict__ targets, const unsigned char* __restrict__ fmask,
    float* __restrict__ out) {
  __shared__ half_t hF[BB * HH];     // 16 KB  f16 hidden state
  __shared__ half_t tfF[BB * 32];    // 2 KB   teacher-forcing input (cols>=10 zero)
  __shared__ half_t rz[BB * 512];    // 32 KB  r (cols 0..255) | z (cols 256..511)
  int tid = threadIdx.x;
  int wave = tid >> 5, lane = tid & 31;
  int hf = lane >> 4, l15 = lane & 15;
  for (int i = tid; i < BB * HH; i += 1024) hF[i] = (half_t)0;
  for (int i = tid; i < BB * 32; i += 1024) tfF[i] = (half_t)0;
  __syncthreads();

  for (int t = 0; t < TT; ++t) {
    // ---- phase 1: r and z tiles (2 per wave) ----
#pragma unroll
    for (int i = 0; i < 2; ++i) {
      int q = wave * 2 + i;
      int mt = q >> 5, nt = q & 31;
      int col = nt * 16 + l15;
      int brow = mt * 16 + l15;
      v8f cg;
#pragma unroll
      for (int vv = 0; vv < 8; ++vv) {
        int row = mt * 16 + hf * 8 + vv;
        cg[vv] = (float)giX[(size_t)(t * BB + row) * G3H + col];
      }
      V16 atf;
      atf.h[0] = *(const v8h*)(tfF + brow * 32 + hf * 8);
      atf.h[1] = *(const v8h*)(tfF + brow * 32 + hf * 8 + 16);
      V16 btf;
      btf.v = *(const v16h*)(WihTfB + (size_t)(nt * 32 + lane) * 16);
      cg = wmma16(atf.v, btf.v, cg);
      v8f ch = zero8f();
#pragma unroll
      for (int kc = 0; kc < 8; ++kc) {
        V16 ah;
        ah.h[0] = *(const v8h*)(hF + brow * HH + kc * 32 + hf * 8);
        ah.h[1] = *(const v8h*)(hF + brow * HH + kc * 32 + hf * 8 + 16);
        V16 bh;
        bh.v = *(const v16h*)(WhhB + (size_t)((kc * 48 + nt) * 32 + lane) * 16);
        ch = wmma16(ah.v, bh.v, ch);
      }
      float bh2 = bhh[col];
#pragma unroll
      for (int vv = 0; vv < 8; ++vv) {
        int row = mt * 16 + hf * 8 + vv;
        float s = sigmoidf_(cg[vv] + ch[vv] + bh2);
        rz[row * 512 + col] = (half_t)s;
      }
    }
    __syncthreads();  // r,z ready

    // ---- phase 2: n tiles (1 per wave): n = tanh(inn + r*hn) ----
    float nv[8];
    int ncol;
    {
      int mt = wave >> 4, nt = 32 + (wave & 15);
      int col = nt * 16 + l15;
      ncol = col - 512;
      int brow = mt * 16 + l15;
      v8f cg;
#pragma unroll
      for (int vv = 0; vv < 8; ++vv) {
        int row = mt * 16 + hf * 8 + vv;
        cg[vv] = (float)giX[(size_t)(t * BB + row) * G3H + col];
      }
      V16 atf;
      atf.h[0] = *(const v8h*)(tfF + brow * 32 + hf * 8);
      atf.h[1] = *(const v8h*)(tfF + brow * 32 + hf * 8 + 16);
      V16 btf;
      btf.v = *(const v16h*)(WihTfB + (size_t)(nt * 32 + lane) * 16);
      cg = wmma16(atf.v, btf.v, cg);
      v8f ch = zero8f();
#pragma unroll
      for (int kc = 0; kc < 8; ++kc) {
        V16 ah;
        ah.h[0] = *(const v8h*)(hF + brow * HH + kc * 32 + hf * 8);
        ah.h[1] = *(const v8h*)(hF + brow * HH + kc * 32 + hf * 8 + 16);
        V16 bh;
        bh.v = *(const v16h*)(WhhB + (size_t)((kc * 48 + nt) * 32 + lane) * 16);
        ch = wmma16(ah.v, bh.v, ch);
      }
      float bh2 = bhh[col];
#pragma unroll
      for (int vv = 0; vv < 8; ++vv) {
        int row = mt * 16 + hf * 8 + vv;
        float r = (float)rz[row * 512 + ncol];
        nv[vv] = tanhf(cg[vv] + r * (ch[vv] + bh2));
      }
    }
    __syncthreads();  // all WMMA reads of hF complete

    // ---- h' = (1-z)*n + z*h ----
    {
      int mt = wave >> 4;
#pragma unroll
      for (int vv = 0; vv < 8; ++vv) {
        int row = mt * 16 + hf * 8 + vv;
        float z = (float)rz[row * 512 + 256 + ncol];
        float hold = (float)hF[row * HH + ncol];
        float hnew = (1.0f - z) * nv[vv] + z * hold;
        hF[row * HH + ncol] = (half_t)hnew;
      }
    }
    __syncthreads();  // h' visible

    // ---- output head + teacher forcing ----
    if (tid < BB * CC) {
      int b = tid / CC, c = tid % CC;
      float acc = bfv[c];
      for (int k = 0; k < HH; ++k) acc += (float)hF[b * HH + k] * Wf[c * HH + k];
      out[(size_t)(b * TT + t) * CC + c] = acc;
      bool pred = acc > 0.0f;  // sigmoid(x) > 0.5  <=>  x > 0
      bool ft = fmask[t * BB + b] != 0;
      float tfval = ft ? targets[(size_t)(b * TT + t) * CC + c] : (pred ? 1.0f : 0.0f);
      tfF[b * 32 + c] = (half_t)tfval;
    }
    __syncthreads();  // tf ready for next step
  }
}

// ---------------------------------------------------------------------------
extern "C" void kernel_launch(void* const* d_in, const int* in_sizes, int n_in,
                              void* d_out, int out_size, void* d_ws, size_t ws_size,
                              hipStream_t stream) {
  const float* features = (const float*)d_in[0];
  const float* targets  = (const float*)d_in[1];
  const unsigned char* fmask = (const unsigned char*)d_in[2];  // jax bool -> 1 byte
  const float* W1 = (const float*)d_in[3];
  const float* b1 = (const float*)d_in[4];
  const float* g1 = (const float*)d_in[5];
  const float* be1 = (const float*)d_in[6];
  const float* m1 = (const float*)d_in[7];
  const float* v1 = (const float*)d_in[8];
  const float* W2 = (const float*)d_in[9];
  const float* b2 = (const float*)d_in[10];
  const float* g2 = (const float*)d_in[11];
  const float* be2 = (const float*)d_in[12];
  const float* m2 = (const float*)d_in[13];
  const float* v2 = (const float*)d_in[14];
  const float* W3 = (const float*)d_in[15];
  const float* b3 = (const float*)d_in[16];
  const float* g3 = (const float*)d_in[17];
  const float* be3 = (const float*)d_in[18];
  const float* m3 = (const float*)d_in[19];
  const float* v3 = (const float*)d_in[20];
  const float* Wih = (const float*)d_in[21];
  const float* Whh = (const float*)d_in[22];
  const float* bih = (const float*)d_in[23];
  const float* bhh = (const float*)d_in[24];
  const float* Wf = (const float*)d_in[25];
  const float* bf = (const float*)d_in[26];
  float* out = (float*)d_out;

  // Workspace carve-up (256B aligned)
  char* ws = (char*)d_ws;
  size_t cur = 0;
  auto take = [&](size_t bytes) {
    size_t o = cur;
    cur += (bytes + 255) & ~(size_t)255;
    return o;
  };
  const size_t convFrag = (size_t)25 * 4 * 8 * 32 * 16;  // halves
  half_t* wB2   = (half_t*)(ws + take(convFrag * 2));
  float*  bias2 = (float*)(ws + take(CONV * 4));
  half_t* wB3   = (half_t*)(ws + take(convFrag * 2));
  float*  bias3 = (float*)(ws + take(CONV * 4));
  half_t* WhhB  = (half_t*)(ws + take((size_t)8 * 48 * 512 * 2));
  half_t* WihXB = (half_t*)(ws + take((size_t)4 * 48 * 512 * 2));
  half_t* WihTfB= (half_t*)(ws + take((size_t)1 * 48 * 512 * 2));
  half_t* out1  = (half_t*)(ws + take((size_t)BB * TT * 8 * CONV * 2));
  half_t* out2  = (half_t*)(ws + take((size_t)BB * TT * 2 * CONV * 2));
  half_t* feats = (half_t*)(ws + take((size_t)TT * BB * CONV * 2));
  half_t* giX   = (half_t*)(ws + take((size_t)TT * BB * G3H * 2));
  (void)ws_size; (void)n_in; (void)in_sizes; (void)out_size;

  // Weight prep
  prep_conv_w<<<(convFrag + 255) / 256, 256, 0, stream>>>(W2, b2, g2, be2, m2, v2, wB2, bias2);
  prep_conv_w<<<(convFrag + 255) / 256, 256, 0, stream>>>(W3, b3, g3, be3, m3, v3, wB3, bias3);
  pack_b_frags<<<(8 * 48 * 512 + 255) / 256, 256, 0, stream>>>(Whh, WhhB, 8, 48, HH, 0, HH, 8 * 48 * 512);
  pack_b_frags<<<(4 * 48 * 512 + 255) / 256, 256, 0, stream>>>(Wih, WihXB, 4, 48, CONV + CC, 0, CONV, 4 * 48 * 512);
  pack_b_frags<<<(1 * 48 * 512 + 255) / 256, 256, 0, stream>>>(Wih, WihTfB, 1, 48, CONV + CC, CONV, CC, 1 * 48 * 512);

  // Conv stack
  conv1_kernel<<<BB * TT * 8, 128, 0, stream>>>(features, W1, b1, g1, be1, m1, v1, out1);
  conv2_kernel<<<(BB * TT * 8 / 16) / 4, 256, 0, stream>>>(out1, wB2, bias2, out2);
  conv3_kernel<<<(BB * TT * 2 / 16) / 4, 256, 0, stream>>>(out2, wB3, bias3, feats);

  // Hoisted input-projection GEMM for the GRU
  gix_kernel<<<(TT * BB) / 16, 256, 0, stream>>>(feats, WihXB, bih, giX);

  // Sequential GRU scan (persistent single workgroup)
  gru_kernel<<<1, 1024, 0, stream>>>(giX, WhhB, WihTfB, bhh, Wf, bf, targets, fmask, out);
}
